// GAT_86535001080291
// MI455X (gfx1250) — matrix operand order repaired and checked by reading, not compiled, not run on hardware
//
#include <hip/hip_runtime.h>
#include <math.h>

typedef __attribute__((ext_vector_type(16))) _Float16 v16h;
typedef __attribute__((ext_vector_type(8)))  float    v8f;

#define NEG_SLOPE 0.2f

__device__ __forceinline__ void atomicMaxFloat(float* addr, float val) {
  // Sign-aware max: positive values via signed int max, negative via unsigned min.
  if (val >= 0.0f) atomicMax((int*)addr, __float_as_int(val));
  else             atomicMin((unsigned int*)addr, (unsigned int)__float_as_int(val));
}

// ---------------- init helpers ----------------
__global__ void fill_kernel(float* p, float v, long n) {
  long i = (long)blockIdx.x * blockDim.x + threadIdx.x;
  long stride = (long)gridDim.x * blockDim.x;
  for (; i < n; i += stride) p[i] = v;
}

__global__ void init_out_kernel(float* __restrict__ out, const float* __restrict__ b,
                                int N, int C) {
  long i = (long)blockIdx.x * blockDim.x + threadIdx.x;
  long n = (long)N * C;
  long stride = (long)gridDim.x * blockDim.x;
  for (; i < n; i += stride) out[i] = b[i % C];
}

// ---------------- WMMA GEMM: out[N,NC] = A[N,64] @ W[64,NC] ----------------
// Each wave computes a 16-row slab across all NC columns using
// v_wmma_f32_16x16x32_f16 (two K=32 chunks).
//  - W is staged once per block into LDS (coalesced b128 global reads);
//    per-lane B fragments then come from conflict-free ds_load_b32.
//  - A fragments are loaded as float4 (global_load_b128) and packed to f16.
template <int NC>
__global__ __launch_bounds__(256) void gemm_wmma_kernel(const float* __restrict__ A,
                                                        const float* __restrict__ W,
                                                        float* __restrict__ out,
                                                        int Nrows) {
  __shared__ float Wlds[64 * NC];

  // Cooperative, coalesced stage of W into LDS (before any wave exits).
  {
    const float4* Wv = (const float4*)W;
    float4* Lv = (float4*)Wlds;
    for (int idx = threadIdx.x; idx < (64 * NC) / 4; idx += blockDim.x)
      Lv[idx] = Wv[idx];
  }
  __syncthreads();

  const int lane = threadIdx.x & 31;
  const int wave = threadIdx.x >> 5;
  const int hi   = lane >> 4;   // which 16-lane half
  const int l16  = lane & 15;
  const int rowBase = (blockIdx.x * (blockDim.x >> 5) + wave) * 16;
  if (rowBase >= Nrows) return;  // wave-uniform exit: EXEC stays all-ones for WMMA

  // A tile: lane holds row M = rowBase + l16.
  // 16-bit A 16x32 layout: element j -> K = j + (j>=8 ? 8:0) + (hi ? 8:0)
  // => per chunk, two contiguous 8-float runs: [hi*8, hi*8+8) and [16+hi*8, 16+hi*8+8)
  v16h a0, a1;
  {
    int row  = rowBase + l16;
    int rowc = row < Nrows ? row : (Nrows - 1);   // clamp; garbage rows never stored
    const float4* ap4 = (const float4*)(A + (long)rowc * 64);
    const int kb = hi * 2;  // hi*8 floats = 2 float4
#pragma unroll
    for (int c = 0; c < 2; ++c) {  // K chunks 0..31 / 32..63
      float4 q0 = ap4[c * 8 + kb + 0];
      float4 q1 = ap4[c * 8 + kb + 1];
      float4 q2 = ap4[c * 8 + 4 + kb + 0];
      float4 q3 = ap4[c * 8 + 4 + kb + 1];
      v16h& a = c ? a1 : a0;
      a[0]  = (_Float16)q0.x; a[1]  = (_Float16)q0.y;
      a[2]  = (_Float16)q0.z; a[3]  = (_Float16)q0.w;
      a[4]  = (_Float16)q1.x; a[5]  = (_Float16)q1.y;
      a[6]  = (_Float16)q1.z; a[7]  = (_Float16)q1.w;
      a[8]  = (_Float16)q2.x; a[9]  = (_Float16)q2.y;
      a[10] = (_Float16)q2.z; a[11] = (_Float16)q2.w;
      a[12] = (_Float16)q3.x; a[13] = (_Float16)q3.y;
      a[14] = (_Float16)q3.z; a[15] = (_Float16)q3.w;
    }
  }

  v8f acc[NC / 16];
#pragma unroll
  for (int t = 0; t < NC / 16; ++t)
    acc[t] = (v8f){0.f, 0.f, 0.f, 0.f, 0.f, 0.f, 0.f, 0.f};

#pragma unroll
  for (int t = 0; t < NC / 16; ++t) {
    // B tile from LDS: lane holds column N = l16; element j -> K = j + (j>=8 ? 8:0) + (hi ? 8:0).
    // Lanes 0..15 read consecutive dwords of one W row -> conflict-free ds_load_b32.
    v16h b0, b1;
#pragma unroll
    for (int j = 0; j < 16; ++j) {
      int k = j + (j >= 8 ? 8 : 0) + (hi ? 8 : 0);
      b0[j] = (_Float16)Wlds[k * NC + t * 16 + l16];
      b1[j] = (_Float16)Wlds[(32 + k) * NC + t * 16 + l16];
    }
    acc[t] = __builtin_amdgcn_wmma_f32_16x16x32_f16(false, a0, false, b0,
                                                    (short)0, acc[t], false, false);
    acc[t] = __builtin_amdgcn_wmma_f32_16x16x32_f16(false, a1, false, b1,
                                                    (short)0, acc[t], false, false);
  }

  // D layout: VGPR r -> row rowBase + r + 8*hi, column l16.
#pragma unroll
  for (int t = 0; t < NC / 16; ++t) {
#pragma unroll
    for (int r = 0; r < 8; ++r) {
      int m = rowBase + r + hi * 8;
      if (m < Nrows) out[(long)m * NC + t * 16 + l16] = acc[t][r];
    }
  }
}

// ---------------- per-node attention coefficients ----------------
// out[n*H+hd] = sum_c h[n, hd*C + c] * a[hd*C + c]
__global__ void alpha_kernel(const float* __restrict__ h, const float* __restrict__ a,
                             float* __restrict__ out, int N, int H, int C) {
  int tid = blockIdx.x * blockDim.x + threadIdx.x;
  if (tid >= N * H) return;
  int n = tid / H, hd = tid % H;
  const float* hp = h + (long)n * H * C + (long)hd * C;
  const float* ap = a + (long)hd * C;
  float s = 0.f;
  for (int c = 0; c < C; ++c) s += hp[c] * ap[c];
  out[tid] = s;
}

// ---------------- edge passes ----------------
__global__ void edge_max_kernel(const int* __restrict__ src, const int* __restrict__ dst,
                                int E, int N, int H,
                                const float* __restrict__ asrc,
                                const float* __restrict__ adst,
                                float* __restrict__ m) {
  long tid = (long)blockIdx.x * blockDim.x + threadIdx.x;
  long total = (long)(E + N) * H;
  if (tid >= total) return;
  int i = (int)(tid / H), hd = (int)(tid % H);
  int s = (i < E) ? src[i] : (i - E);
  int d = (i < E) ? dst[i] : (i - E);
  float e = asrc[s * H + hd] + adst[d * H + hd];
  e = e > 0.f ? e : NEG_SLOPE * e;
  atomicMaxFloat(&m[d * H + hd], e);
}

__global__ void edge_denom_kernel(const int* __restrict__ src, const int* __restrict__ dst,
                                  int E, int N, int H,
                                  const float* __restrict__ asrc,
                                  const float* __restrict__ adst,
                                  const float* __restrict__ m,
                                  float* __restrict__ den) {
  long tid = (long)blockIdx.x * blockDim.x + threadIdx.x;
  long total = (long)(E + N) * H;
  if (tid >= total) return;
  int i = (int)(tid / H), hd = (int)(tid % H);
  int s = (i < E) ? src[i] : (i - E);
  int d = (i < E) ? dst[i] : (i - E);
  float e = asrc[s * H + hd] + adst[d * H + hd];
  e = e > 0.f ? e : NEG_SLOPE * e;
  atomicAdd(&den[d * H + hd], expf(e - m[d * H + hd]));
}

// One edge per C-thread sub-group (256/C edges per block); lane = feature column -> coalesced
// gather of h[src] and coalesced f32 atomic adds to agg[dst].
template <int C, int H>
__global__ __launch_bounds__(256) void edge_agg_kernel(const int* __restrict__ src,
                                                       const int* __restrict__ dst,
                                                       int E, int N,
                                                       const float* __restrict__ asrc,
                                                       const float* __restrict__ adst,
                                                       const float* __restrict__ m,
                                                       const float* __restrict__ den,
                                                       const float* __restrict__ h,
                                                       float* __restrict__ agg) {
  const int EPB = 256 / C;
  int sub = threadIdx.x / C;
  int col = threadIdx.x % C;
  long i = (long)blockIdx.x * EPB + sub;
  if (i >= (long)E + N) return;
  int s = (i < (long)E) ? src[i] : (int)(i - E);
  int d = (i < (long)E) ? dst[i] : (int)(i - E);
  int hd = (H == 1) ? 0 : (col >> 5);  // 32 channels per head
  float e = asrc[s * H + hd] + adst[d * H + hd];
  e = e > 0.f ? e : NEG_SLOPE * e;
  float alpha = expf(e - m[d * H + hd]) / den[d * H + hd];
  atomicAdd(&agg[(long)d * C + col], h[(long)s * C + col] * alpha);
}

// ---------------- fused bias + exact gelu (in place) ----------------
__global__ void gelu_bias_kernel(float* __restrict__ x, const float* __restrict__ b,
                                 int N, int C) {
  long i = (long)blockIdx.x * blockDim.x + threadIdx.x;
  long n = (long)N * C;
  long stride = (long)gridDim.x * blockDim.x;
  for (; i < n; i += stride) {
    float v = x[i] + b[i % C];
    x[i] = 0.5f * v * (1.0f + erff(v * 0.70710678118654752f));
  }
}

extern "C" void kernel_launch(void* const* d_in, const int* in_sizes, int n_in,
                              void* d_out, int out_size, void* d_ws, size_t ws_size,
                              hipStream_t stream) {
  const float* x    = (const float*)d_in[0];
  const int*   ei   = (const int*)d_in[1];
  const float* W1   = (const float*)d_in[2];
  const float* aS1  = (const float*)d_in[3];
  const float* aD1  = (const float*)d_in[4];
  const float* b1   = (const float*)d_in[5];
  const float* W2   = (const float*)d_in[6];
  const float* aS2  = (const float*)d_in[7];
  const float* aD2  = (const float*)d_in[8];
  const float* b2   = (const float*)d_in[9];
  float* out = (float*)d_out;

  const int N = in_sizes[0] / 64;
  const int E = in_sizes[1] / 2;
  const int* srcIdx = ei;       // edge_index[0]
  const int* dstIdx = ei + E;   // edge_index[1]

  // workspace layout (floats): 172*N total (~69 MB)
  float* ws   = (float*)d_ws;
  float* h1   = ws; ws += (long)N * 64;   // layer-1 features
  float* agg1 = ws; ws += (long)N * 64;   // layer-1 aggregate -> becomes h2 after gelu
  float* g2   = ws; ws += (long)N * 32;   // layer-2 features
  float* as1  = ws; ws += 2L * N;
  float* ad1  = ws; ws += 2L * N;
  float* m1   = ws; ws += 2L * N;
  float* dn1  = ws; ws += 2L * N;
  float* as2  = ws; ws += (long)N;
  float* ad2  = ws; ws += (long)N;
  float* m2   = ws; ws += (long)N;
  float* dn2  = ws; ws += (long)N;

  const float NEG_INF = -__builtin_inff();

  // ---- init (every call; harness does not re-zero) ----
  fill_kernel<<<2048, 256, 0, stream>>>(m1, NEG_INF, 2L * N);
  fill_kernel<<<2048, 256, 0, stream>>>(dn1, 0.f, 2L * N);
  fill_kernel<<<4096, 256, 0, stream>>>(agg1, 0.f, 64L * N);
  fill_kernel<<<2048, 256, 0, stream>>>(m2, NEG_INF, (long)N);
  fill_kernel<<<2048, 256, 0, stream>>>(dn2, 0.f, (long)N);
  init_out_kernel<<<4096, 256, 0, stream>>>(out, b2, N, 32);  // out = b2 + segsum

  // ---- layer 1 ----
  {
    int blocks = (N + 127) / 128;  // 8 waves/block * 16 rows
    gemm_wmma_kernel<64><<<blocks, 256, 0, stream>>>(x, W1, h1, N);
  }
  {
    int total = N * 2;
    alpha_kernel<<<(total + 255) / 256, 256, 0, stream>>>(h1, aS1, as1, N, 2, 32);
    alpha_kernel<<<(total + 255) / 256, 256, 0, stream>>>(h1, aD1, ad1, N, 2, 32);
  }
  {
    long total = (long)(E + N) * 2;
    int blocks = (int)((total + 255) / 256);
    edge_max_kernel<<<blocks, 256, 0, stream>>>(srcIdx, dstIdx, E, N, 2, as1, ad1, m1);
    edge_denom_kernel<<<blocks, 256, 0, stream>>>(srcIdx, dstIdx, E, N, 2, as1, ad1, m1, dn1);
    int ablocks = (int)(((long)E + N + 3) / 4);  // 4 edges per 256-thread block
    edge_agg_kernel<64, 2><<<ablocks, 256, 0, stream>>>(srcIdx, dstIdx, E, N,
                                                        as1, ad1, m1, dn1, h1, agg1);
  }
  gelu_bias_kernel<<<4096, 256, 0, stream>>>(agg1, b1, N, 64);  // agg1 -> h2 in place

  // ---- layer 2 ----
  {
    int blocks = (N + 127) / 128;
    gemm_wmma_kernel<32><<<blocks, 256, 0, stream>>>(agg1, W2, g2, N);
  }
  {
    alpha_kernel<<<(N + 255) / 256, 256, 0, stream>>>(g2, aS2, as2, N, 1, 32);
    alpha_kernel<<<(N + 255) / 256, 256, 0, stream>>>(g2, aD2, ad2, N, 1, 32);
  }
  {
    long total = (long)(E + N);
    int blocks = (int)((total + 255) / 256);
    edge_max_kernel<<<blocks, 256, 0, stream>>>(srcIdx, dstIdx, E, N, 1, as2, ad2, m2);
    edge_denom_kernel<<<blocks, 256, 0, stream>>>(srcIdx, dstIdx, E, N, 1, as2, ad2, m2, dn2);
    int ablocks = (int)(((long)E + N + 7) / 8);  // 8 edges per block
    edge_agg_kernel<32, 1><<<ablocks, 256, 0, stream>>>(srcIdx, dstIdx, E, N,
                                                        as2, ad2, m2, dn2, g2, out);
  }
}
